// DWAModel_23794118819891
// MI455X (gfx1250) — matrix operand order repaired and checked by reading, not compiled, not run on hardware
//
#include <hip/hip_runtime.h>
#include <hip/hip_bf16.h>
#include <math.h>

// ---------------------------------------------------------------------------
// CDNA5 (gfx1250, wave32) implementation of the DWA model.
//   - bf16 WMMA (v_wmma_f32_16x16x32_bf16) for all large GEMMs
//   - f32 WMMA  (v_wmma_f32_16x16x4_f32)   for the per-n U@V rank-8 products
//   - fused W_assembled kernel: double-buffered LDS pipeline, alpha tiles
//     staged with global_load_async_to_lds_b128 (ASYNCcnt / s_wait_asynccnt).
// ---------------------------------------------------------------------------

typedef __bf16 bf16_t;
typedef __attribute__((ext_vector_type(16))) __bf16 v16bf;
typedef __attribute__((ext_vector_type(8)))  float  v8f;
typedef __attribute__((ext_vector_type(2)))  float  v2f;

// Problem dimensions (compile-time constants of the reference).
static constexpr int BATCH = 256;
static constexpr int DMODEL = 512;
static constexpr int HID  = 1024;
static constexpr int DA   = 256;
static constexpr int DB   = 256;
static constexpr int RNK  = 8;
static constexpr int NPOOL = 2048;
static constexpr int DPOOL = 4608;
static constexpr int SASP = 4;
static constexpr int DKEY = 64;
static constexpr int AQ   = SASP * DKEY;         // 256
static constexpr int UEND = DB * RNK;            // 2048
static constexpr int VEND = UEND + RNK * DA;     // 4096

__device__ __forceinline__ float gelu_tanh(float x) {
  const float k0 = 0.7978845608028654f;  // sqrt(2/pi)
  float x3 = x * x * x;
  return 0.5f * x * (1.0f + tanhf(k0 * (x + 0.044715f * x3)));
}

// --- CDNA5 async global->LDS copy (tracked by ASYNCcnt) --------------------
__device__ __forceinline__ void async_load_b128(unsigned lds_byte_addr,
                                                const void* gaddr) {
  asm volatile("global_load_async_to_lds_b128 %0, %1, off"
               :: "v"(lds_byte_addr), "v"(gaddr)
               : "memory");
}
__device__ __forceinline__ void wait_asynccnt0() {
  asm volatile("s_wait_asynccnt 0" ::: "memory");
}

// ---------------------------------------------------------------------------
// Pack helpers: f32 -> bf16 (row-major, optional source row stride / gather)
// ---------------------------------------------------------------------------
__global__ __launch_bounds__(256) void pack_bf16(
    const float* __restrict__ src, bf16_t* __restrict__ dst,
    int total, int cols, int srcStride) {
  int i = blockIdx.x * 256 + threadIdx.x;
  if (i >= total) return;
  int r = i / cols, c = i - r * cols;
  dst[i] = (bf16_t)src[(size_t)r * srcStride + c];
}

// dst[k][s*DK + q] = src[s][k][q]   (aspect-concat of W_Q / W_K)
__global__ __launch_bounds__(256) void pack_cat_bf16(
    const float* __restrict__ src, bf16_t* __restrict__ dst,
    int Krows, int S, int DK) {
  int i = blockIdx.x * 256 + threadIdx.x;
  int total = Krows * S * DK;
  if (i >= total) return;
  int cols = S * DK;
  int k = i / cols, c = i - k * cols;
  int s = c / DK, q = c - s * DK;
  dst[i] = (bf16_t)src[((size_t)s * Krows + k) * DK + q];
}

// ---------------------------------------------------------------------------
// Generic bf16-WMMA GEMM: C[M,N] = epi(A[M,K] * B[K,N] + bias[N])
//   A: f32 row-major (converted to bf16 on the fly), B: bf16 row-major.
//   Workgroup tile 128x64, 8 waves (4x2), wave tile 32x32 (2x2 WMMA tiles).
//   M % 128 == 0, N % 64 == 0, K % 32 == 0 (true for every call site).
// A-fragment layout (16-bit A 16x32): lanes 0-15 = rows, K runs {0-7,16-23};
//   lanes 16-31 same rows, K runs {8-15,24-31}  -> two contiguous 8-f32 runs.
// B-fragment layout (16-bit B 32x16): lane holds column lane%16, same K runs as A
//   -> 16 bf16 loads strided by N.
// ---------------------------------------------------------------------------
__global__ __launch_bounds__(256) void gemm_bf16_wmma(
    const float* __restrict__ A, const bf16_t* __restrict__ B,
    const float* __restrict__ bias, float* __restrict__ C,
    int M, int N, int K, int epi) {
  const int wave  = threadIdx.x >> 5;
  const int lane  = threadIdx.x & 31;
  const int lrow  = lane & 15;
  const int lhalf = lane >> 4;
  const int m_base = blockIdx.x * 128 + (wave >> 1) * 32;
  const int n_base = blockIdx.y * 64 + (wave & 1) * 32;

  v8f acc[2][2] = {};

  for (int k0 = 0; k0 < K; k0 += 32) {
    // A fragments (f32 -> bf16)
    v16bf af[2];
#pragma unroll
    for (int i = 0; i < 2; ++i) {
      const float* ap =
          A + (size_t)(m_base + 16 * i + lrow) * K + k0 + (lhalf ? 8 : 0);
      float4 f0 = *(const float4*)(ap);
      float4 f1 = *(const float4*)(ap + 4);
      float4 f2 = *(const float4*)(ap + 16);
      float4 f3 = *(const float4*)(ap + 20);
      v16bf a;
      a[0] = (bf16_t)f0.x;  a[1] = (bf16_t)f0.y;  a[2] = (bf16_t)f0.z;  a[3] = (bf16_t)f0.w;
      a[4] = (bf16_t)f1.x;  a[5] = (bf16_t)f1.y;  a[6] = (bf16_t)f1.z;  a[7] = (bf16_t)f1.w;
      a[8] = (bf16_t)f2.x;  a[9] = (bf16_t)f2.y;  a[10] = (bf16_t)f2.z; a[11] = (bf16_t)f2.w;
      a[12] = (bf16_t)f3.x; a[13] = (bf16_t)f3.y; a[14] = (bf16_t)f3.z; a[15] = (bf16_t)f3.w;
      af[i] = a;
    }
    // B fragments: lane col = lane%16, K runs {0-7,16-23} / {8-15,24-31} per half
    v16bf bfr[2];
#pragma unroll
    for (int j = 0; j < 2; ++j) {
      const bf16_t* bp = B + (size_t)(k0 + 8 * lhalf) * N + n_base + 16 * j + lrow;
      v16bf b;
#pragma unroll
      for (int s = 0; s < 8; ++s) { b[s] = bp[(size_t)s * N]; b[8 + s] = bp[(size_t)(16 + s) * N]; }
      bfr[j] = b;
    }
    if (k0 + 32 < K)  // prefetch next B K-slice (global_prefetch_b8)
      __builtin_prefetch((const void*)(B + (size_t)(k0 + 32 + lane) * N + n_base), 0, 0);
#pragma unroll
    for (int i = 0; i < 2; ++i)
#pragma unroll
      for (int j = 0; j < 2; ++j)
        acc[i][j] = __builtin_amdgcn_wmma_f32_16x16x32_bf16(
            false, af[i], false, bfr[j], (short)0, acc[i][j], false, false);
  }

  // Epilogue. C/D layout: lane col = lane%16; reg p -> row p + 8*(lane>=16).
#pragma unroll
  for (int j = 0; j < 2; ++j) {
    const int n = n_base + 16 * j + lrow;
    const float bv = bias ? bias[n] : 0.0f;
#pragma unroll
    for (int i = 0; i < 2; ++i) {
#pragma unroll
      for (int p = 0; p < 8; ++p) {
        const int m = m_base + 16 * i + p + 8 * lhalf;
        float v = acc[i][j][p] + bv;
        if (epi == 1) v = gelu_tanh(v);
        C[(size_t)m * N + n] = v;
      }
    }
  }
}

// ---------------------------------------------------------------------------
// Normalize keys (per n, per aspect) and store transposed bf16 [aq][n]
// (that transposed layout is exactly the B matrix of the scores GEMM).
// ---------------------------------------------------------------------------
__global__ __launch_bounds__(256) void knorm_kernel(
    const float* __restrict__ keys, bf16_t* __restrict__ knormT) {
  const int n = blockIdx.x;       // 0..2047
  const int t = threadIdx.x;      // aq index 0..255
  float v = keys[(size_t)n * AQ + t];
  __shared__ float red[256];
  red[t] = v * v;
  __syncthreads();
  for (int s = 32; s >= 1; s >>= 1) {
    if ((t & 63) < s) red[t] += red[t + s];
    __syncthreads();
  }
  float nrm = sqrtf(red[t & ~63]);
  knormT[(size_t)t * NPOOL + n] = (bf16_t)(v / (nrm + 1e-8f));
}

// Normalize queries, fold in softmax(aspect_logits) weight; f32 [b][aq].
__global__ __launch_bounds__(256) void qnorm_kernel(
    const float* __restrict__ q, const float* __restrict__ logits,
    float* __restrict__ qw) {
  const int b = blockIdx.x;
  const int t = threadIdx.x;
  float v = q[(size_t)b * AQ + t];
  float l0 = logits[0], l1 = logits[1], l2 = logits[2], l3 = logits[3];
  float mx = fmaxf(fmaxf(l0, l1), fmaxf(l2, l3));
  float e0 = expf(l0 - mx), e1 = expf(l1 - mx), e2 = expf(l2 - mx), e3 = expf(l3 - mx);
  float es = e0 + e1 + e2 + e3;
  int a = t >> 6;
  float wa = ((a == 0) ? e0 : (a == 1) ? e1 : (a == 2) ? e2 : e3) / es;
  __shared__ float red[256];
  red[t] = v * v;
  __syncthreads();
  for (int s = 32; s >= 1; s >>= 1) {
    if ((t & 63) < s) red[t] += red[t + s];
    __syncthreads();
  }
  float nrm = sqrtf(red[t & ~63]);
  qw[(size_t)b * AQ + t] = wa * v / (nrm + 1e-8f);
}

// scores -> alpha (gated, exp, row-normalized over n). One block per batch row.
__global__ __launch_bounds__(256) void alpha_kernel(
    const float* __restrict__ scores, const float* __restrict__ tau_p,
    float* __restrict__ alpha) {
  const int b = blockIdx.x;
  const int t = threadIdx.x;
  const float tau = tau_p[0];
  float raw[8];
  float loc = 0.0f;
#pragma unroll
  for (int kk = 0; kk < 8; ++kk) {
    float s = scores[(size_t)b * NPOOL + t + kk * 256];
    float g = 1.0f / (1.0f + expf(-(s - tau)));   // LAMBDA_SHARP = 1
    float r = g * expf(s);                         // TEMPERATURE = 1
    raw[kk] = r;
    loc += r;
  }
  __shared__ float red[256];
  red[t] = loc;
  __syncthreads();
  for (int s = 128; s >= 1; s >>= 1) {
    if (t < s) red[t] += red[t + s];
    __syncthreads();
  }
  float inv = 1.0f / (red[0] + 1e-8f);
#pragma unroll
  for (int kk = 0; kk < 8; ++kk)
    alpha[(size_t)b * NPOOL + t + kk * 256] = raw[kk] * inv;
}

// ---------------------------------------------------------------------------
// Fused W_assembled:
//   W[b,d,e] = W_base[d,e] + sum_n alpha[b,n] * (U[n] @ V[n])[d,e]
// Grid: 1024 workgroups = 4 b-blocks (64 rows) x 256 (d,e) 16x16 tiles.
// Double-buffered software pipeline over 32-n K-slices:
//   producer: alpha tile -> LDS via global_load_async_to_lds_b128 (ASYNCcnt)
//             UV tiles   -> f32 WMMA (r=8 -> two K=4 steps) -> bf16 in LDS
//   consumer: 4 alpha A-fragments (ds loads) x 2 d_local columns per wave of
//             v_wmma_f32_16x16x32_bf16.
// One s_wait_asynccnt + one barrier per slice; producer of slice s+1 overlaps
// consumer of slice s.
// ---------------------------------------------------------------------------
__global__ __launch_bounds__(256) void assemble_W_kernel(
    const float* __restrict__ pool, const float* __restrict__ alpha,
    const float* __restrict__ W_base, float* __restrict__ Wout) {
  const int b0 = (blockIdx.x >> 8) * 64;
  const int tile = blockIdx.x & 255;
  const int d0 = (tile >> 4) * 16;
  const int e0 = (tile & 15) * 16;
  const int wave = threadIdx.x >> 5;
  const int lane = threadIdx.x & 31;
  const int lrow = lane & 15;
  const int lhalf = lane >> 4;

  __shared__ __align__(32) bf16_t uvbf[2][32 * 256];  // [buf][n_local][d_local*16+e]
  __shared__ __align__(16) float salpha[2][64 * 32];  // [buf][row][n_local]

  v8f acc[4][2] = {};

  // ---- producer: stage alpha tile for slice starting at n0 into buf ----
  auto stage_alpha = [&](int buf, int n0) {
    const int row = threadIdx.x >> 2;          // 0..63
    const int c8  = (threadIdx.x & 3) * 8;     // 0,8,16,24
    const float* src = alpha + (size_t)(b0 + row) * NPOOL + n0 + c8;
    unsigned dst = (unsigned)(size_t)&salpha[buf][row * 32 + c8];
    async_load_b128(dst, (const void*)src);
    async_load_b128(dst + 16, (const void*)(src + 4));
  };

  // ---- producer: UV tiles via f32 WMMA (wave w owns n_local 4w..4w+3) ----
  auto produce_uv = [&](int buf, int n0) {
#pragma unroll
    for (int nn = 0; nn < 4; ++nn) {
      const int nl = (wave << 2) + nn;
      const int n = n0 + nl;
      const float* Un = pool + (size_t)n * DPOOL;        // U[n][d][r] = Un[d*8+r]
      const float* Vn = Un + UEND;                        // V[n][r][e] = Vn[r*256+e]
      v8f uv = {};
#pragma unroll
      for (int s = 0; s < 2; ++s) {
        // f32 A 16x4: lanes 0-15 rows, K {r0,r0+1}; lanes 16-31 rows, K {r0+2,r0+3}
        const int rA = 4 * s + 2 * lhalf;
        const float* up = Un + (size_t)(d0 + lrow) * RNK + rA;
        v2f a;  a[0] = up[0]; a[1] = up[1];
        v2f bv; bv[0] = Vn[(size_t)rA * DA + e0 + lrow];
                bv[1] = Vn[(size_t)(rA + 1) * DA + e0 + lrow];
        uv = __builtin_amdgcn_wmma_f32_16x16x4_f32(
            false, a, false, bv, (short)0, uv, false, false);
      }
      // D layout: col = lrow, row = p + 8*lhalf  -> LDS B matrix for consumer
#pragma unroll
      for (int p = 0; p < 8; ++p) {
        const int dl = p + 8 * lhalf;
        uvbf[buf][nl * 256 + dl * 16 + lrow] = (bf16_t)uv[p];
      }
    }
  };

  // ---- consumer: main bf16 WMMA contraction over the staged slice ----
  auto consume = [&](int buf) {
    v16bf af[4];
#pragma unroll
    for (int i = 0; i < 4; ++i) {
      const float* ap = &salpha[buf][(16 * i + lrow) * 32 + (lhalf ? 8 : 0)];
      float4 f0 = *(const float4*)(ap);
      float4 f1 = *(const float4*)(ap + 4);
      float4 f2 = *(const float4*)(ap + 16);
      float4 f3 = *(const float4*)(ap + 20);
      v16bf a;
      a[0] = (bf16_t)f0.x;  a[1] = (bf16_t)f0.y;  a[2] = (bf16_t)f0.z;  a[3] = (bf16_t)f0.w;
      a[4] = (bf16_t)f1.x;  a[5] = (bf16_t)f1.y;  a[6] = (bf16_t)f1.z;  a[7] = (bf16_t)f1.w;
      a[8] = (bf16_t)f2.x;  a[9] = (bf16_t)f2.y;  a[10] = (bf16_t)f2.z; a[11] = (bf16_t)f2.w;
      a[12] = (bf16_t)f3.x; a[13] = (bf16_t)f3.y; a[14] = (bf16_t)f3.z; a[15] = (bf16_t)f3.w;
      af[i] = a;
    }
#pragma unroll
    for (int jj = 0; jj < 2; ++jj) {
      const int j = wave * 2 + jj;  // d_local column block this wave owns
      v16bf bfrag;
#pragma unroll
      for (int s = 0; s < 8; ++s) {
        bfrag[s] = uvbf[buf][(8 * lhalf + s) * 256 + j * 16 + lrow];
        bfrag[8 + s] = uvbf[buf][(16 + 8 * lhalf + s) * 256 + j * 16 + lrow];
      }
#pragma unroll
      for (int i = 0; i < 4; ++i)
        acc[i][jj] = __builtin_amdgcn_wmma_f32_16x16x32_bf16(
            false, af[i], false, bfrag, (short)0, acc[i][jj], false, false);
    }
  };

  constexpr int NS = NPOOL / 32;  // 64 slices

  // Prologue: fill buffer 0.
  stage_alpha(0, 0);
  produce_uv(0, 0);
  wait_asynccnt0();
  __syncthreads();

  for (int s = 0; s < NS; ++s) {
    const int cur = s & 1;
    if (s + 1 < NS) {                 // produce next slice while consuming cur
      stage_alpha(1 - cur, (s + 1) * 32);
      produce_uv(1 - cur, (s + 1) * 32);
    }
    consume(cur);
    wait_asynccnt0();                 // drain this wave's async LDS writes
    __syncthreads();                  // publish buffers workgroup-wide
  }

  // Epilogue: add W_base, scatter to W_assembled[b, d, e].
#pragma unroll
  for (int jj = 0; jj < 2; ++jj) {
    const int d = d0 + wave * 2 + jj;
    const int e = e0 + lrow;
    const float wb = W_base[(size_t)d * DA + e];
#pragma unroll
    for (int i = 0; i < 4; ++i) {
#pragma unroll
      for (int p = 0; p < 8; ++p) {
        const int b = b0 + 16 * i + p + 8 * lhalf;
        Wout[(size_t)b * (DB * DA) + (size_t)d * DA + e] = acc[i][jj][p] + wb;
      }
    }
  }
}

// ---------------------------------------------------------------------------
// Fused h_t = einsum('ba,bca->bc', h_A, W_asm) + b_asm ; y = h_A + gamma*h_t ;
// LayerNorm(y) -> h_mid.  One block (256 threads) per batch row.
// ---------------------------------------------------------------------------
__global__ __launch_bounds__(256) void ht_ln_kernel(
    const float* __restrict__ hA, const float* __restrict__ Wasm,
    const float* __restrict__ basm, const float* __restrict__ gamma_p,
    const float* __restrict__ ln_scale, const float* __restrict__ ln_bias,
    float* __restrict__ hmid) {
  const int b = blockIdx.x;
  const int c = threadIdx.x;
  __shared__ float sA[256];
  __shared__ float red[256];
  sA[c] = hA[(size_t)b * DA + c];
  __syncthreads();
  const float4* wr = (const float4*)(Wasm + (size_t)b * (DB * DA) + (size_t)c * DA);
  float dot = 0.0f;
#pragma unroll 8
  for (int a4 = 0; a4 < DA / 4; ++a4) {
    float4 w = wr[a4];
    const float4 h = *(const float4*)(sA + a4 * 4);
    dot += w.x * h.x + w.y * h.y + w.z * h.z + w.w * h.w;
  }
  float ht = dot + basm[(size_t)b * DB + c];
  float y = sA[c] + gamma_p[0] * ht;
  red[c] = y;
  __syncthreads();
  for (int s = 128; s >= 1; s >>= 1) {
    if (c < s) red[c] += red[c + s];
    __syncthreads();
  }
  float mu = red[0] * (1.0f / 256.0f);
  __syncthreads();
  float dy = y - mu;
  red[c] = dy * dy;
  __syncthreads();
  for (int s = 128; s >= 1; s >>= 1) {
    if (c < s) red[c] += red[c + s];
    __syncthreads();
  }
  float var = red[0] * (1.0f / 256.0f);
  float inv = rsqrtf(var + 1e-6f);
  hmid[(size_t)b * DA + c] = dy * inv * ln_scale[c] + ln_bias[c];
}

// ---------------------------------------------------------------------------
// Host orchestration
// ---------------------------------------------------------------------------
extern "C" void kernel_launch(void* const* d_in, const int* in_sizes, int n_in,
                              void* d_out, int out_size, void* d_ws, size_t ws_size,
                              hipStream_t stream) {
  (void)in_sizes; (void)n_in; (void)out_size; (void)ws_size;

  const float* x        = (const float*)d_in[0];
  const float* pool     = (const float*)d_in[1];
  const float* A_w0     = (const float*)d_in[2];
  const float* A_b0     = (const float*)d_in[3];
  const float* A_w1     = (const float*)d_in[4];
  const float* A_b1     = (const float*)d_in[5];
  const float* W_Q      = (const float*)d_in[6];
  const float* W_K      = (const float*)d_in[7];
  const float* logits   = (const float*)d_in[8];
  const float* tau      = (const float*)d_in[9];
  const float* W_base   = (const float*)d_in[10];
  const float* b_base   = (const float*)d_in[11];
  const float* gamma    = (const float*)d_in[12];
  const float* ln_scale = (const float*)d_in[13];
  const float* ln_bias  = (const float*)d_in[14];
  const float* B_w0     = (const float*)d_in[15];
  const float* B_b0     = (const float*)d_in[16];
  const float* B_w1     = (const float*)d_in[17];
  const float* B_b1     = (const float*)d_in[18];

  // Output regions (tuple order: output, alpha, W_assembled, keys)
  float* out_y  = (float*)d_out;
  float* alpha  = out_y + (size_t)BATCH * DMODEL;
  float* Wasm   = alpha + (size_t)BATCH * NPOOL;
  float* keys   = Wasm + (size_t)BATCH * DB * DA;

  // Workspace carve (256-byte aligned slabs, ~13 MB total)
  char* p = (char*)d_ws;
  auto carve = [&](size_t bytes) -> char* {
    char* r = p;
    p += (bytes + 255) & ~(size_t)255;
    return r;
  };
  float*  h1     = (float*)carve((size_t)BATCH * HID * 4);
  float*  hA     = (float*)carve((size_t)BATCH * DA * 4);
  float*  qbuf   = (float*)carve((size_t)BATCH * AQ * 4);
  float*  qw     = (float*)carve((size_t)BATCH * AQ * 4);
  float*  scores = (float*)carve((size_t)BATCH * NPOOL * 4);
  float*  basm   = (float*)carve((size_t)BATCH * DB * 4);
  float*  hmid   = (float*)carve((size_t)BATCH * DA * 4);
  float*  t1     = (float*)carve((size_t)BATCH * HID * 4);
  bf16_t* Aw0b   = (bf16_t*)carve((size_t)DMODEL * HID * 2);
  bf16_t* Aw1b   = (bf16_t*)carve((size_t)HID * DA * 2);
  bf16_t* WQb    = (bf16_t*)carve((size_t)DA * AQ * 2);
  bf16_t* WKb    = (bf16_t*)carve((size_t)DPOOL * AQ * 2);
  bf16_t* Bw0b   = (bf16_t*)carve((size_t)DA * HID * 2);
  bf16_t* Bw1b   = (bf16_t*)carve((size_t)HID * DMODEL * 2);
  bf16_t* biasb  = (bf16_t*)carve((size_t)NPOOL * DB * 2);
  bf16_t* knTb   = (bf16_t*)carve((size_t)AQ * NPOOL * 2);

  const dim3 blk(256);
  auto cdiv = [](int a, int b) { return (a + b - 1) / b; };

  // --- weight packs (bf16) ---
  pack_bf16<<<cdiv(DMODEL * HID, 256), blk, 0, stream>>>(A_w0, Aw0b, DMODEL * HID, HID, HID);
  pack_bf16<<<cdiv(HID * DA, 256), blk, 0, stream>>>(A_w1, Aw1b, HID * DA, DA, DA);
  pack_cat_bf16<<<cdiv(DA * AQ, 256), blk, 0, stream>>>(W_Q, WQb, DA, SASP, DKEY);
  pack_cat_bf16<<<cdiv(DPOOL * AQ, 256), blk, 0, stream>>>(W_K, WKb, DPOOL, SASP, DKEY);
  pack_bf16<<<cdiv(DA * HID, 256), blk, 0, stream>>>(B_w0, Bw0b, DA * HID, HID, HID);
  pack_bf16<<<cdiv(HID * DMODEL, 256), blk, 0, stream>>>(B_w1, Bw1b, HID * DMODEL, DMODEL, DMODEL);
  pack_bf16<<<cdiv(NPOOL * DB, 256), blk, 0, stream>>>(pool + VEND, biasb, NPOOL * DB, DB, DPOOL);

  // --- A-MLP ---
  gemm_bf16_wmma<<<dim3(BATCH / 128, HID / 64), blk, 0, stream>>>(
      x, Aw0b, A_b0, h1, BATCH, HID, DMODEL, 1);
  gemm_bf16_wmma<<<dim3(BATCH / 128, DA / 64), blk, 0, stream>>>(
      h1, Aw1b, A_b1, hA, BATCH, DA, HID, 0);

  // --- queries / keys ---
  gemm_bf16_wmma<<<dim3(BATCH / 128, AQ / 64), blk, 0, stream>>>(
      hA, WQb, nullptr, qbuf, BATCH, AQ, DA, 0);
  gemm_bf16_wmma<<<dim3(NPOOL / 128, AQ / 64), blk, 0, stream>>>(
      pool, WKb, nullptr, keys, NPOOL, AQ, DPOOL, 0);

  // --- normalize, scores, alpha ---
  knorm_kernel<<<NPOOL, blk, 0, stream>>>(keys, knTb);
  qnorm_kernel<<<BATCH, blk, 0, stream>>>(qbuf, logits, qw);
  gemm_bf16_wmma<<<dim3(BATCH / 128, NPOOL / 64), blk, 0, stream>>>(
      qw, knTb, nullptr, scores, BATCH, NPOOL, AQ, 0);
  alpha_kernel<<<BATCH, blk, 0, stream>>>(scores, tau, alpha);

  // --- b_assembled = b_base + alpha @ bias ---
  gemm_bf16_wmma<<<dim3(BATCH / 128, DB / 64), blk, 0, stream>>>(
      alpha, biasb, b_base, basm, BATCH, DB, NPOOL, 0);

  // --- fused W_assembled (dominant kernel) ---
  assemble_W_kernel<<<1024, blk, 0, stream>>>(pool, alpha, W_base, Wasm);

  // --- h_t + residual + LayerNorm ---
  ht_ln_kernel<<<BATCH, blk, 0, stream>>>(hA, Wasm, basm, gamma, ln_scale, ln_bias, hmid);

  // --- B-MLP ---
  gemm_bf16_wmma<<<dim3(BATCH / 128, HID / 64), blk, 0, stream>>>(
      hmid, Bw0b, B_b0, t1, BATCH, HID, DA, 1);
  gemm_bf16_wmma<<<dim3(BATCH / 128, DMODEL / 64), blk, 0, stream>>>(
      t1, Bw1b, B_b1, out_y, BATCH, DMODEL, HID, 0);
}
